// FeatureFusionAttention_8031588843741
// MI455X (gfx1250) — compile-verified, matrix operand order checked by reading, not run on hardware
//
#include <hip/hip_runtime.h>
#include <hip/hip_bf16.h>
#include <cstdint>
#include <cstddef>

// ---------------------------------------------------------------------------
// FeatureFusionAttention for MI455X (gfx1250, wave32, WMMA bf16 16x16x32)
//
// Roofline: ~26 GFLOP vs ~400 MB mandatory HBM traffic -> memory bound
// (~17us at 23.3 TB/s) provided matmuls run on the matrix pipe. All six
// matmul stages use v_wmma_f32_16x16x32_bf16; softmax stats in fp32.
// GEMM A-tiles (bf16 path) are staged into LDS by the Tensor Data Mover
// (tensor_load_to_lds + s_wait_tensorcnt), overlapping with W-tile convert.
// ---------------------------------------------------------------------------

typedef __attribute__((ext_vector_type(16))) __bf16 v16bf;
typedef __attribute__((ext_vector_type(8)))  float  v8f;
typedef __attribute__((ext_vector_type(4)))  unsigned int u32x4;
typedef __attribute__((ext_vector_type(8)))  int i32x8;
typedef __attribute__((ext_vector_type(4)))  int i32x4;

#define DEVINL __device__ __forceinline__

#ifndef __has_builtin
#define __has_builtin(x) 0
#endif
#if __has_builtin(__builtin_amdgcn_tensor_load_to_lds) && \
    __has_builtin(__builtin_amdgcn_s_wait_tensorcnt)
#define USE_TDM 1
#else
#define USE_TDM 0
#endif

// problem constants (fixed by the reference)
#define BATCH 32
#define NPIX  784          // 28*28
#define CHI   1024
#define CLO   512
#define DMODEL 128
#define NHEAD 4
#define HDIM  32

DEVINL unsigned short f2bfbits(float f) {
  unsigned u = __builtin_bit_cast(unsigned, f);
  return (unsigned short)((u + 0x7FFFu + ((u >> 16) & 1u)) >> 16);  // RNE
}
DEVINL __bf16 bits2bf(unsigned short s) { return __builtin_bit_cast(__bf16, s); }

// ISA 7.12.2: 16-bit A-matrix 16x32 element map. For vector element e (0..15)
// of the v16bf fragment held by a lane with half-wave `group` (lane>>4):
//   K = 16*(e>=8) + 2*((e>>1)&3) + 8*group + (e&1)
// (two contiguous 8-element runs per lane -> loads vectorize to 2x b128)
DEVINL int amap_k(int e, int group) {
  int jj = e >> 1;
  return ((jj >> 2) << 4) + ((jj & 3) << 1) + (group << 3) + (e & 1);
}

DEVINL v16bf frag_from_row(const unsigned short* rowbase, int group) {
  v16bf f;
#pragma unroll
  for (int e = 0; e < 16; ++e) f[e] = bits2bf(rowbase[amap_k(e, group)]);
  return f;
}

DEVINL v8f wmma_bf16(v16bf a, v16bf b, v8f c) {
  return __builtin_amdgcn_wmma_f32_16x16x32_bf16(
      /*neg_a=*/false, a, /*neg_b=*/false, b,
      /*c_mod=*/(short)0, c, /*reuse_a=*/false, /*reuse_b=*/false);
}

#if USE_TDM
// TDM: DMA a rows x cols tile of bf16 (2-byte) data, row stride
// `row_stride_elems`, from global memory into LDS at byte offset `lds_off`.
// D# bit packing per CDNA5 ISA sec. 8.3 (group0) / 8.4 (group1); groups 2/3
// unused (2D tensor) and passed as zeros. 6-arg builtin (clang-23 toolchain).
DEVINL void tdm_load_2d_bf16(unsigned lds_off, const void* gptr,
                             unsigned rows, unsigned cols,
                             unsigned row_stride_elems) {
  unsigned long long ga = (unsigned long long)(uintptr_t)gptr;
  const unsigned dim0 = row_stride_elems;   // tensor width (OOB bound)
  const unsigned dim1 = 1u << 20;           // tall tensor bound
  u32x4 g0;
  g0[0] = 1u;                                          // count=1, user mode
  g0[1] = lds_off;                                     // lds_addr
  g0[2] = (unsigned)(ga & 0xFFFFFFFFu);                // global_addr[31:0]
  g0[3] = (unsigned)((ga >> 32) & 0x01FFFFFFu) | (2u << 30);  // [56:32]|type=2
  i32x8 g1;
  g1[0] = (int)(1u << 16);                             // data_size = 2 bytes
  g1[1] = (int)((dim0 & 0xFFFFu) << 16);               // tensor_dim0[15:0]
  g1[2] = (int)(((dim0 >> 16) & 0xFFFFu) | ((dim1 & 0xFFFFu) << 16));
  g1[3] = (int)(((dim1 >> 16) & 0xFFFFu) | ((cols & 0xFFFFu) << 16)); // tile_dim0
  g1[4] = (int)(rows & 0xFFFFu);                       // tile_dim1 (tile_dim2=0)
  g1[5] = (int)row_stride_elems;                       // tensor_dim0_stride
  g1[6] = 0;
  g1[7] = 0;
  i32x4 gz4; gz4[0] = 0; gz4[1] = 0; gz4[2] = 0; gz4[3] = 0;
  i32x8 gz8;
#pragma unroll
  for (int i = 0; i < 8; ++i) gz8[i] = 0;
  __builtin_amdgcn_tensor_load_to_lds(g0, g1, gz4, gz4, gz8, 0);
}
#endif

// ---------------------------------------------------------------------------
// Kernel 1: 2x2 avg-pool of low_feat [B,512,56,56] -> bf16 [B,784,512]
// ---------------------------------------------------------------------------
__global__ __launch_bounds__(256) void pool_transpose_low(
    const float* __restrict__ low, unsigned short* __restrict__ lf) {
  __shared__ unsigned short T[32][33];
  const int b  = blockIdx.z;
  const int n0 = blockIdx.x * 32;
  const int c0 = blockIdx.y * 32;
  const int tx = threadIdx.x & 31, ty = threadIdx.x >> 5;  // 32 x 8

#pragma unroll
  for (int i = 0; i < 4; ++i) {
    int c = c0 + ty + i * 8;
    int n = n0 + tx;
    if (n < NPIX) {
      int hh = n / 28, ww = n % 28;
      const float* p = low + (((size_t)b * CLO + c) * 56 + 2 * hh) * 56 + 2 * ww;
      float s = p[0] + p[1] + p[56] + p[57];
      T[ty + i * 8][tx] = f2bfbits(0.25f * s);
    }
  }
  __syncthreads();
#pragma unroll
  for (int i = 0; i < 4; ++i) {
    int n = n0 + ty + i * 8;
    int c = c0 + tx;
    if (n < NPIX)
      lf[((size_t)b * NPIX + n) * CLO + c] = T[tx][ty + i * 8];
  }
}

// ---------------------------------------------------------------------------
// Kernel 2: generic WMMA GEMM with bias.
//   Out[m, c] = sum_k A[m,k] * W[c,k] + bias[c],  M = B*NPIX = 25088
// A_F32_CHAN:   A from fp32 channel-major [Kdim, NPIX] per batch (high_feat);
//               else bf16 row-major [M, Kdim]  -> staged by the TDM.
// OUT_F32_CHAN: store fp32 channel-major [B, Cout, NPIX]; else bf16 [M, Cout].
// Block: 256 thr (8 waves), tile 64 rows x 128 cols, K step 32.
// ---------------------------------------------------------------------------
template <bool A_F32_CHAN, bool OUT_F32_CHAN>
__global__ __launch_bounds__(256) void gemm_bias_kernel(
    const void* __restrict__ Asrc, const float* __restrict__ Wmat,
    const float* __restrict__ bias, void* __restrict__ Out,
    int Kdim, int Cout) {
  __shared__ unsigned short As[64 * 32];   // A tile, K-major rows
  __shared__ unsigned short Bs[128 * 32];  // W tile, K-major rows (per out col)

  const int tid = threadIdx.x;
  const int m0 = blockIdx.x * 64;
  const int c0 = blockIdx.y * 128;
  const int w = tid >> 5, lane = tid & 31;
  const int group = lane >> 4, lc = lane & 15;
  const int mt = w & 3;        // this wave's 16-row tile
  const int ch = w >> 2;       // this wave's 64-col half

  v8f acc[4];
#pragma unroll
  for (int t = 0; t < 4; ++t) acc[t] = v8f{};

  for (int k0 = 0; k0 < Kdim; k0 += 32) {
    // --- stage A (64x32 bf16) ---
    if (A_F32_CHAN) {
      const float* A = (const float*)Asrc;
      for (int idx = tid; idx < 64 * 32; idx += 256) {
        int kk = idx >> 6, mm = idx & 63;            // coalesce along pixels
        int m = m0 + mm;
        int bb = m / NPIX, nn = m % NPIX;
        float v = A[((size_t)bb * Kdim + (k0 + kk)) * NPIX + nn];
        As[mm * 32 + kk] = f2bfbits(v);
      }
    } else {
#if USE_TDM
      if (w == 0) {   // one wave issues the DMA; EXEC ignored by TDM
        const unsigned short* gsrc =
            (const unsigned short*)Asrc + (size_t)m0 * Kdim + k0;
        tdm_load_2d_bf16((unsigned)(uintptr_t)(void*)As, gsrc,
                         /*rows=*/64, /*cols=*/32, (unsigned)Kdim);
      }
#else
      const unsigned short* A = (const unsigned short*)Asrc;
      for (int idx = tid; idx < 64 * 32; idx += 256) {
        int mm = idx >> 5, kk = idx & 31;            // coalesce along K
        As[mm * 32 + kk] = A[(size_t)(m0 + mm) * Kdim + (k0 + kk)];
      }
#endif
    }
    // --- stage W (128x32 bf16), overlaps with the TDM transfer ---
    for (int idx = tid; idx < 128 * 32; idx += 256) {
      int cc = idx >> 5, kk = idx & 31;
      Bs[cc * 32 + kk] = f2bfbits(Wmat[(size_t)(c0 + cc) * Kdim + (k0 + kk)]);
    }
#if USE_TDM
    if (!A_F32_CHAN && w == 0) __builtin_amdgcn_s_wait_tensorcnt(0);
#endif
    __syncthreads();

    v16bf af = frag_from_row(As + (mt * 16 + lc) * 32, group);
#pragma unroll
    for (int t = 0; t < 4; ++t) {
      v16bf bf = frag_from_row(Bs + (ch * 64 + t * 16 + lc) * 32, group);
      acc[t] = wmma_bf16(af, bf, acc[t]);
    }
    __syncthreads();
  }

  // --- epilogue: bias + store (C layout: lane holds col lc, rows v+8*group) ---
#pragma unroll
  for (int t = 0; t < 4; ++t) {
    int c = c0 + ch * 64 + t * 16 + lc;
    float bv = bias[c];
#pragma unroll
    for (int v = 0; v < 8; ++v) {
      int m = m0 + mt * 16 + v + 8 * group;
      float val = acc[t][v] + bv;
      if (OUT_F32_CHAN) {
        int bb = m / NPIX, nn = m % NPIX;
        ((float*)Out)[((size_t)bb * Cout + c) * NPIX + nn] = val;
      } else {
        ((unsigned short*)Out)[(size_t)m * Cout + c] = f2bfbits(val);
      }
    }
  }
}

// ---------------------------------------------------------------------------
// Kernel 3: flash attention. One wave = one 16-query tile of one (b, head).
// N = 784 = 49 tiles of 16; key loop in chunks of 32 (WMMA K dim).
// Q,K,V,fused are bf16 [B*784, 128] with head h at cols [32h, 32h+32).
// ---------------------------------------------------------------------------
__global__ __launch_bounds__(128) void attention_kernel(
    const unsigned short* __restrict__ Qb, const unsigned short* __restrict__ Kb,
    const unsigned short* __restrict__ Vb, unsigned short* __restrict__ Fused) {
  __shared__ unsigned short Pls[4][16 * 32];  // per-wave P tile (q x key)

  const int w = threadIdx.x >> 5, lane = threadIdx.x & 31;
  const int group = lane >> 4, lc = lane & 15;
  const int bh = blockIdx.x;           // b*NHEAD + h
  const int b = bh >> 2, h = bh & 3;
  const int qt = blockIdx.y * 4 + w;
  if (qt >= 49) return;
  const int q0 = qt * 16;

  const size_t base = (size_t)b * NPIX * DMODEL + (size_t)h * HDIM;
  const unsigned short* Qp = Qb + base;
  const unsigned short* Kp = Kb + base;
  const unsigned short* Vp = Vb + base;
  unsigned short* Pw = Pls[w];

  // Q fragment: A-matrix 16x32, row = q0 + lc
  v16bf qa;
#pragma unroll
  for (int e = 0; e < 16; ++e)
    qa[e] = bits2bf(Qp[(size_t)(q0 + lc) * DMODEL + amap_k(e, group)]);

  const float scale = 0.17677669529663689f;  // 32^-0.5
  float mrow[8], srow[8];
#pragma unroll
  for (int v = 0; v < 8; ++v) { mrow[v] = -1e30f; srow[v] = 0.0f; }
  v8f o0 = v8f{}, o1 = v8f{};

  for (int kc = 0; kc < NPIX; kc += 32) {
    const bool has2 = (kc + 16) < NPIX;  // uniform

    // scores: Q(16x32) x K^T(32x16) per 16-key tile
    v16bf kb0;
#pragma unroll
    for (int e = 0; e < 16; ++e)
      kb0[e] = bits2bf(Kp[(size_t)(kc + lc) * DMODEL + amap_k(e, group)]);
    v8f s0 = wmma_bf16(qa, kb0, v8f{});
    v8f s1 = v8f{};
    if (has2) {
      v16bf kb1;
#pragma unroll
      for (int e = 0; e < 16; ++e)
        kb1[e] = bits2bf(Kp[(size_t)(kc + 16 + lc) * DMODEL + amap_k(e, group)]);
      s1 = wmma_bf16(qa, kb1, v8f{});
    }

    // online softmax; row r = v + 8*group lives across the 16-lane half
#pragma unroll
    for (int v = 0; v < 8; ++v) {
      float x0 = s0[v] * scale;
      float x1 = has2 ? s1[v] * scale : -1e30f;
      float mx = fmaxf(x0, x1);
#pragma unroll
      for (int off = 1; off < 16; off <<= 1)
        mx = fmaxf(mx, __shfl_xor(mx, off, 32));
      float mn = fmaxf(mrow[v], mx);
      float alpha = __expf(mrow[v] - mn);
      float p0 = __expf(x0 - mn);
      float p1 = has2 ? __expf(x1 - mn) : 0.0f;
      float ps = p0 + p1;
#pragma unroll
      for (int off = 1; off < 16; off <<= 1)
        ps += __shfl_xor(ps, off, 32);
      srow[v] = srow[v] * alpha + ps;
      mrow[v] = mn;
      o0[v] *= alpha;
      o1[v] *= alpha;
      // stash probabilities (row = v + 8*group, key col = lc / 16+lc)
      Pw[(v + 8 * group) * 32 + lc]      = f2bfbits(p0);
      Pw[(v + 8 * group) * 32 + 16 + lc] = f2bfbits(p1);
    }

    // P(16x32) x V(32x16) for both halves of head_dim
    v16bf pa = frag_from_row(Pw + lc * 32, group);
    v16bf vb0, vb1;
#pragma unroll
    for (int e = 0; e < 16; ++e) {
      int key = kc + amap_k(e, group);
      if (key > NPIX - 1) key = NPIX - 1;  // P is zero there anyway
      vb0[e] = bits2bf(Vp[(size_t)key * DMODEL + lc]);
      vb1[e] = bits2bf(Vp[(size_t)key * DMODEL + 16 + lc]);
    }
    o0 = wmma_bf16(pa, vb0, o0);
    o1 = wmma_bf16(pa, vb1, o1);
  }

  // normalize + store fused bf16 [B*N, 128]; v_rcp_f32 instead of full divide
#pragma unroll
  for (int v = 0; v < 8; ++v) {
    float inv = __builtin_amdgcn_rcpf(srow[v]);
    size_t row = (size_t)b * NPIX + q0 + v + 8 * group;
    Fused[row * DMODEL + h * HDIM + lc]      = f2bfbits(o0[v] * inv);
    Fused[row * DMODEL + h * HDIM + 16 + lc] = f2bfbits(o1[v] * inv);
  }
}

// ---------------------------------------------------------------------------
// Launch
// ---------------------------------------------------------------------------
extern "C" void kernel_launch(void* const* d_in, const int* in_sizes, int n_in,
                              void* d_out, int out_size, void* d_ws, size_t ws_size,
                              hipStream_t stream) {
  const float* high = (const float*)d_in[0];  // [32,1024,28,28]
  const float* low  = (const float*)d_in[1];  // [32,512,56,56]
  const float* Wq = (const float*)d_in[2];
  const float* bq = (const float*)d_in[3];
  const float* Wk = (const float*)d_in[4];
  const float* bk = (const float*)d_in[5];
  const float* Wv = (const float*)d_in[6];
  const float* bv = (const float*)d_in[7];
  const float* Wo = (const float*)d_in[8];
  const float* bo = (const float*)d_in[9];

  const size_t M = (size_t)BATCH * NPIX;           // 25088
  unsigned short* lf    = (unsigned short*)d_ws;                     // [M,512]
  unsigned short* Qbuf  = lf    + M * CLO;                           // [M,128]
  unsigned short* Kbuf  = Qbuf  + M * DMODEL;
  unsigned short* Vbuf  = Kbuf  + M * DMODEL;
  unsigned short* Fused = Vbuf  + M * DMODEL;
  (void)in_sizes; (void)n_in; (void)out_size; (void)ws_size;

  // 1) pool + transpose low_feat -> bf16 [B,784,512]
  pool_transpose_low<<<dim3(25, 16, BATCH), 256, 0, stream>>>(low, lf);

  // 2) projections (M x 128), WMMA bf16
  gemm_bias_kernel<true, false><<<dim3(392, 1), 256, 0, stream>>>(
      high, Wq, bq, Qbuf, CHI, DMODEL);                 // Q from channel-major fp32
  gemm_bias_kernel<false, false><<<dim3(392, 1), 256, 0, stream>>>(
      lf, Wk, bk, Kbuf, CLO, DMODEL);
  gemm_bias_kernel<false, false><<<dim3(392, 1), 256, 0, stream>>>(
      lf, Wv, bv, Vbuf, CLO, DMODEL);

  // 3) flash attention per (b, head, 16-query tile)
  attention_kernel<<<dim3(BATCH * NHEAD, 13), 128, 0, stream>>>(
      Qbuf, Kbuf, Vbuf, Fused);

  // 4) output projection, store fp32 directly as [B,512,28,28]
  gemm_bias_kernel<false, true><<<dim3(392, 4), 256, 0, stream>>>(
      Fused, Wo, bo, (float*)d_out, DMODEL, CLO);
}
